// CCSABlockWithLiteMLA_55602646614390
// MI455X (gfx1250) — compile-verified, matrix operand order checked by reading, not run on hardware
//
#include <hip/hip_runtime.h>
#include <hip/hip_bf16.h>

typedef __attribute__((ext_vector_type(16))) _Float16 v16h;
typedef __attribute__((ext_vector_type(8)))  _Float16 v8h;
typedef __attribute__((ext_vector_type(8)))  float    v8f;

// ---------------- problem constants ----------------
constexpr int B_   = 32;
constexpr int H_   = 64;
constexpr int W_   = 64;
constexpr int N_   = H_ * W_;      // 4096
constexpr int CIN_ = 960;
constexpr int C3_  = 96;           // qkv channels
constexpr int NG_  = 12;           // attention groups
constexpr int F1_  = 192, F2_ = 256, F3_ = 512;

// ---------------- workspace layout (bytes) ----------------
constexpr size_t XH_OFF  = 0;                                   // LN'd input, f16 [B][N][960]
constexpr size_t XH_SZ   = (size_t)B_ * N_ * CIN_ * 2;
constexpr size_t WQ_OFF  = XH_OFF + XH_SZ;                      // qkv_w f16 [96][960]
constexpr size_t WQ_SZ   = (size_t)C3_ * CIN_ * 2;
constexpr size_t WP_OFF  = WQ_OFF + WQ_SZ;                      // proj_w f16 [960][96]
constexpr size_t WP_SZ   = (size_t)CIN_ * C3_ * 2;
constexpr size_t QKV_OFF = WP_OFF + WP_SZ;                      // qkv f16 [B][96][N]
constexpr size_t QKV_SZ  = (size_t)B_ * C3_ * N_ * 2;
constexpr size_t M3_OFF  = QKV_OFF + QKV_SZ;                    // m3 f16 [B][96][N]
constexpr size_t M5_OFF  = M3_OFF + QKV_SZ;                     // m5 f16 [B][96][N]
constexpr size_t VK_OFF  = M5_OFF + QKV_SZ;                     // vk f32 [B][12][9][8]
constexpr size_t VK_SZ   = (size_t)B_ * NG_ * 72 * 4;
constexpr size_t ATT_OFF = VK_OFF + VK_SZ;                      // att f16 [B][N][96]

// Build a 16-half WMMA operand from two contiguous 8-half runs (p, p+16).
__device__ __forceinline__ v16h load_tile16(const _Float16* p) {
  v8h lo = *reinterpret_cast<const v8h*>(p);
  v8h hi = *reinterpret_cast<const v8h*>(p + 16);
  return __builtin_shufflevector(lo, hi, 0,1,2,3,4,5,6,7,8,9,10,11,12,13,14,15);
}

// LDS byte address for async-copy destination (low 32 bits of the flat address).
__device__ __forceinline__ unsigned lds_addr32(const void* p) {
  return (unsigned)(uintptr_t)p;
}

// CDNA5 async global->LDS 16-byte copy (ASYNCcnt-tracked).
__device__ __forceinline__ void async_copy_b128(unsigned lds_dst, const void* gsrc) {
  asm volatile("global_load_async_to_lds_b128 %0, %1, off"
               :: "v"(lds_dst), "v"(gsrc) : "memory");
}

// ---------------- 0) weight conversion f32 -> f16 ----------------
__global__ void cvt_weights(const float* __restrict__ qkv_w,
                            const float* __restrict__ proj_w,
                            _Float16* __restrict__ wq,
                            _Float16* __restrict__ wp) {
  int i = blockIdx.x * blockDim.x + threadIdx.x;
  if (i < C3_ * CIN_) {
    wq[i] = (_Float16)qkv_w[i];     // [96][960] row-major, K contiguous
    wp[i] = (_Float16)proj_w[i];    // [960][96] row-major, K contiguous
  }
}

// ---------------- 1) LayerNorm -> f16 [B][N][960] (K contiguous) ----------------
__global__ void ln_kernel(const float* __restrict__ x1, const float* __restrict__ g1, const float* __restrict__ b1,
                          const float* __restrict__ x2, const float* __restrict__ g2, const float* __restrict__ b2,
                          const float* __restrict__ x3, const float* __restrict__ g3, const float* __restrict__ b3,
                          _Float16* __restrict__ xh) {
  const int lane = threadIdx.x & 31;
  const int wid  = blockIdx.x * (blockDim.x >> 5) + (threadIdx.x >> 5);
  const int total = B_ * N_ * 3;
  if (wid >= total) return;
  const int seg = wid % 3;
  const int row = wid / 3;
  const int b   = row / N_;
  const int n   = row % N_;

  int width, base;
  const float *xp, *gp, *bp;
  if (seg == 0)      { width = F1_; base = 0;         xp = x1; gp = g1; bp = b1; }
  else if (seg == 1) { width = F2_; base = F1_;       xp = x2; gp = g2; bp = b2; }
  else               { width = F3_; base = F1_ + F2_; xp = x3; gp = g3; bp = b3; }

  const float* xrow = xp + ((size_t)b * N_ + n) * width;
  const int cnt = width >> 5;               // 6 / 8 / 16
  float vals[16];
  float s = 0.f, ss = 0.f;
  #pragma unroll
  for (int i = 0; i < 16; ++i) {
    if (i < cnt) {
      float v = xrow[lane + (i << 5)];
      vals[i] = v; s += v; ss += v * v;
    }
  }
  #pragma unroll
  for (int o = 16; o >= 1; o >>= 1) { s += __shfl_xor(s, o); ss += __shfl_xor(ss, o); }
  const float inv  = 1.f / (float)width;
  const float mean = s * inv;
  const float var  = ss * inv - mean * mean;
  const float rs   = rsqrtf(var + 1e-6f);

  _Float16* orow = xh + ((size_t)b * N_ + n) * CIN_ + base;
  #pragma unroll
  for (int i = 0; i < 16; ++i) {
    if (i < cnt) {
      int c = lane + (i << 5);
      orow[c] = (_Float16)((vals[i] - mean) * rs * gp[c] + bp[c]);
    }
  }
}

// ---------------- 2) qkv GEMM: [96x960] @ [960xN] via WMMA ----------------
// Block = 4 waves sharing the weight panel; K chunked 15 x 64, async
// double-buffered into LDS. Each wave owns one 16-wide N tile (6 accumulators).
// All 6 A tiles are preloaded so the 6 WMMAs issue back-to-back.
__global__ void __launch_bounds__(128) qkv_gemm(const _Float16* __restrict__ xh,
                                                const _Float16* __restrict__ wq,
                                                _Float16* __restrict__ qkvh) {
  __shared__ _Float16 wbuf[2][C3_ * 64];          // 2 x 12 KB

  const int tid  = threadIdx.x;
  const int lane = tid & 31;
  const int wv   = tid >> 5;
  const int b  = blockIdx.x >> 6;                 // 64 n-groups per batch
  const int nt = ((blockIdx.x & 63) << 2) + wv;   // 4 tiles per block
  const int n0 = nt << 4;

  const int mrow = lane & 15;
  const int kb   = (lane >> 4) << 3;              // 0 or 8
  const _Float16* xrow = xh + ((size_t)b * N_ + (n0 + mrow)) * CIN_ + kb;

  // Stage one 96x64 weight chunk: 768 x 16B transfers, 6 per thread.
  auto stage = [&](int c) {
    #pragma unroll
    for (int t = 0; t < 6; ++t) {
      int idx  = tid + (t << 7);
      int m    = idx >> 3;
      int part = idx & 7;
      async_copy_b128(lds_addr32(&wbuf[c & 1][(m << 6) + (part << 3)]),
                      wq + (size_t)m * CIN_ + c * 64 + (part << 3));
    }
  };

  v8f acc[6] = {};
  stage(0);
  for (int c = 0; c < 15; ++c) {
    if (c < 14) {
      stage(c + 1);                                       // prefetch next chunk
      asm volatile("s_wait_asynccnt 0x6" ::: "memory");   // chunk c complete
    } else {
      asm volatile("s_wait_asynccnt 0x0" ::: "memory");
    }
    __syncthreads();
    const _Float16* wc = &wbuf[c & 1][0];
    #pragma unroll
    for (int kk = 0; kk < 2; ++kk) {
      const int k0 = kk << 5;
      v16h bt = load_tile16(xrow + c * 64 + k0);          // global B tile first
      v16h a[6];
      #pragma unroll
      for (int m = 0; m < 6; ++m)                         // 12 ds_load_b128, one wait
        a[m] = load_tile16(wc + (((m << 4) + mrow) << 6) + k0 + kb);
      #pragma unroll
      for (int m = 0; m < 6; ++m)                         // WMMAs back-to-back
        acc[m] = __builtin_amdgcn_wmma_f32_16x16x32_f16(false, a[m], false, bt,
                                                        (short)0, acc[m], false, false);
    }
    __syncthreads();   // all waves done with wbuf[c&1] before it is re-staged
  }

  // C/D layout: lane gives N, vgpr r gives M=r (lanes 0-15) or r+8 (lanes 16-31)
  const int nc = n0 + (lane & 15);
  const int mo = (lane < 16) ? 0 : 8;
  #pragma unroll
  for (int m = 0; m < 6; ++m)
    #pragma unroll
    for (int r = 0; r < 8; ++r) {
      int c = (m << 4) + mo + r;
      qkvh[((size_t)b * C3_ + c) * N_ + nc] = (_Float16)acc[m][r];
    }
}

// ---------------- 3) fused dw3+pw3 and dw5+pw5 per 8-channel group ----------------
__global__ void msconv(const _Float16* __restrict__ qkvh,
                       const float* __restrict__ dw3, const float* __restrict__ pw3,
                       const float* __restrict__ dw5, const float* __restrict__ pw5,
                       _Float16* __restrict__ m3h, _Float16* __restrict__ m5h) {
  __shared__ float tile[8][20][20];     // 16x16 pixels + halo 2
  __shared__ float wdw3[8][9], wdw5[8][25], wpw3[8][8], wpw5[8][8];

  const int tid = threadIdx.x;
  const int idx = blockIdx.x;
  const int b  = idx / (NG_ * 16);
  const int r0 = idx % (NG_ * 16);
  const int j  = r0 >> 4;               // pw group (12 groups of 8 ch)
  const int t  = r0 & 15;
  const int ty = (t >> 2) << 4;
  const int tx = (t & 3) << 4;

  for (int i = tid; i < 8 * 20 * 20; i += 256) {
    int c  = i / 400, p = i % 400, yy = p / 20, xx = p % 20;
    int gy = ty + yy - 2, gx = tx + xx - 2;
    float v = 0.f;
    if (gy >= 0 && gy < H_ && gx >= 0 && gx < W_)
      v = (float)qkvh[((size_t)b * C3_ + (j * 8 + c)) * N_ + gy * W_ + gx];
    tile[c][yy][xx] = v;
  }
  for (int i = tid; i < 8 * 9;  i += 256) wdw3[i / 9][i % 9]   = dw3[(j * 8 + i / 9) * 9 + i % 9];
  for (int i = tid; i < 8 * 25; i += 256) wdw5[i / 25][i % 25] = dw5[(j * 8 + i / 25) * 25 + i % 25];
  for (int i = tid; i < 64;     i += 256) { wpw3[i >> 3][i & 7] = pw3[(j * 8 + (i >> 3)) * 8 + (i & 7)];
                                            wpw5[i >> 3][i & 7] = pw5[(j * 8 + (i >> 3)) * 8 + (i & 7)]; }
  __syncthreads();

  const int y = tid >> 4, x = tid & 15;
  const int cy = y + 2, cx = x + 2;
  float d3[8], d5[8];
  #pragma unroll
  for (int i = 0; i < 8; ++i) {
    float s3 = 0.f;
    #pragma unroll
    for (int dy = -1; dy <= 1; ++dy)
      #pragma unroll
      for (int dx = -1; dx <= 1; ++dx)
        s3 += wdw3[i][(dy + 1) * 3 + dx + 1] * tile[i][cy + dy][cx + dx];
    float s5 = 0.f;
    #pragma unroll
    for (int dy = -2; dy <= 2; ++dy)
      #pragma unroll
      for (int dx = -2; dx <= 2; ++dx)
        s5 += wdw5[i][(dy + 2) * 5 + dx + 2] * tile[i][cy + dy][cx + dx];
    d3[i] = s3; d5[i] = s5;
  }
  const size_t pix = (size_t)(ty + y) * W_ + (tx + x);
  #pragma unroll
  for (int o = 0; o < 8; ++o) {
    float a3 = 0.f, a5 = 0.f;
    #pragma unroll
    for (int i = 0; i < 8; ++i) { a3 += wpw3[o][i] * d3[i]; a5 += wpw5[o][i] * d5[i]; }
    size_t off = ((size_t)b * C3_ + (j * 8 + o)) * N_ + pix;
    m3h[off] = (_Float16)a3;
    m5h[off] = (_Float16)a5;
  }
}

// ---------------- 4) vk[9][8] reduction per (b, g) ----------------
__global__ void vk_reduce(const _Float16* __restrict__ qkvh,
                          const _Float16* __restrict__ m3h,
                          const _Float16* __restrict__ m5h,
                          float* __restrict__ vkout) {
  const int b = blockIdx.x / NG_;
  const int g = blockIdx.x % NG_;
  const _Float16* S = (g < 4) ? qkvh : (g < 8) ? m3h : m5h;
  const int base = (g & 3) * 24;
  const _Float16* kp = S + ((size_t)b * C3_ + base + 8)  * N_;
  const _Float16* vp = S + ((size_t)b * C3_ + base + 16) * N_;

  float acc[9][8] = {};
  for (int n = threadIdx.x; n < N_; n += 256) {
    float kv[8], vv[8];
    #pragma unroll
    for (int e = 0; e < 8; ++e) kv[e] = fmaxf((float)kp[(size_t)e * N_ + n], 0.f);
    #pragma unroll
    for (int d = 0; d < 8; ++d) vv[d] = (float)vp[(size_t)d * N_ + n];
    #pragma unroll
    for (int d = 0; d < 8; ++d)
      #pragma unroll
      for (int e = 0; e < 8; ++e) acc[d][e] += vv[d] * kv[e];
    #pragma unroll
    for (int e = 0; e < 8; ++e) acc[8][e] += kv[e];
  }
  #pragma unroll
  for (int d = 0; d < 9; ++d)
    #pragma unroll
    for (int e = 0; e < 8; ++e)
      #pragma unroll
      for (int o = 16; o >= 1; o >>= 1) acc[d][e] += __shfl_xor(acc[d][e], o);

  __shared__ float vkLds[72];
  if (threadIdx.x < 72) vkLds[threadIdx.x] = 0.f;
  __syncthreads();
  if ((threadIdx.x & 31) == 0) {
    #pragma unroll
    for (int d = 0; d < 9; ++d)
      #pragma unroll
      for (int e = 0; e < 8; ++e) atomicAdd(&vkLds[d * 8 + e], acc[d][e]);  // ds_add_f32
  }
  __syncthreads();
  if (threadIdx.x < 72) vkout[(size_t)blockIdx.x * 72 + threadIdx.x] = vkLds[threadIdx.x];
}

// ---------------- 5) att = (vk @ relu(q)) with denominator; f16 [B][N][96] ----------------
__global__ void att_apply(const _Float16* __restrict__ qkvh,
                          const _Float16* __restrict__ m3h,
                          const _Float16* __restrict__ m5h,
                          const float* __restrict__ vk,
                          _Float16* __restrict__ atth) {
  const int idx = blockIdx.x;
  const int b = idx / (NG_ * 16);
  const int r = idx % (NG_ * 16);
  const int g = r >> 4;
  const int n = ((r & 15) << 8) + threadIdx.x;

  __shared__ float vkS[72];
  if (threadIdx.x < 72) vkS[threadIdx.x] = vk[(size_t)(b * NG_ + g) * 72 + threadIdx.x];
  __syncthreads();

  const _Float16* S = (g < 4) ? qkvh : (g < 8) ? m3h : m5h;
  const int base = (g & 3) * 24;
  const _Float16* qp = S + ((size_t)b * C3_ + base) * N_;

  float q[8];
  #pragma unroll
  for (int e = 0; e < 8; ++e) q[e] = fmaxf((float)qp[(size_t)e * N_ + n], 0.f);
  float den = 0.f;
  #pragma unroll
  for (int e = 0; e < 8; ++e) den += vkS[64 + e] * q[e];
  const float inv = 1.f / (den + 1e-15f);

  v8h outv;
  #pragma unroll
  for (int d = 0; d < 8; ++d) {
    float num = 0.f;
    #pragma unroll
    for (int e = 0; e < 8; ++e) num += vkS[d * 8 + e] * q[e];
    outv[d] = (_Float16)(num * inv);
  }
  *reinterpret_cast<v8h*>(atth + ((size_t)b * N_ + n) * C3_ + g * 8) = outv;
}

// ---------------- 6) proj GEMM [960x96]@[96xN] + BN + residual + split ----------------
// Block = 4 waves sharing one async-staged 96x96 weight panel in LDS.
__global__ void __launch_bounds__(128) proj_gemm(const _Float16* __restrict__ atth,
                          const _Float16* __restrict__ wp,
                          const float* __restrict__ bn_g, const float* __restrict__ bn_b,
                          const float* __restrict__ bn_m, const float* __restrict__ bn_v,
                          const float* __restrict__ x1, const float* __restrict__ x2,
                          const float* __restrict__ x3,
                          float* __restrict__ out1, float* __restrict__ out2,
                          float* __restrict__ out3) {
  __shared__ _Float16 wpanel[C3_ * C3_];          // 18 KB

  const int tid  = threadIdx.x;
  const int lane = tid & 31;
  const int wv   = tid >> 5;
  const int mb = blockIdx.x % 10;                 // 10 panels of 96 rows -> 960
  const int ng = (blockIdx.x / 10) & 63;
  const int b  = blockIdx.x / 640;
  const int nt = (ng << 2) + wv;
  const int n0 = nt << 4;
  const int mbase = mb * 96;

  // Stage 96x96 panel: 1152 x 16B transfers, 9 per thread.
  #pragma unroll
  for (int t = 0; t < 9; ++t) {
    int idx  = tid + (t << 7);
    int m    = idx / 12;
    int part = idx % 12;
    async_copy_b128(lds_addr32(&wpanel[m * C3_ + part * 8]),
                    wp + (size_t)(mbase + m) * C3_ + part * 8);
  }
  asm volatile("s_wait_asynccnt 0x0" ::: "memory");
  __syncthreads();

  const int mrow = lane & 15;
  const int kb   = (lane >> 4) << 3;
  const _Float16* arow = atth + ((size_t)b * N_ + (n0 + mrow)) * C3_ + kb;

  v8f acc[6] = {};
  #pragma unroll
  for (int k0 = 0; k0 < C3_; k0 += 32) {
    v16h bt = load_tile16(arow + k0);               // global B tile first
    v16h a[6];
    #pragma unroll
    for (int m = 0; m < 6; ++m)
      a[m] = load_tile16(&wpanel[((m << 4) + mrow) * C3_ + k0 + kb]);
    #pragma unroll
    for (int m = 0; m < 6; ++m)                     // WMMAs back-to-back
      acc[m] = __builtin_amdgcn_wmma_f32_16x16x32_f16(false, a[m], false, bt,
                                                      (short)0, acc[m], false, false);
  }

  const int nc = n0 + (lane & 15);
  const int mo = (lane < 16) ? 0 : 8;
  #pragma unroll
  for (int m = 0; m < 6; ++m) {
    const int m0 = mbase + (m << 4);           // 16-row tile never crosses 192/448
    int width, cb; const float* xin; float* outp;
    if (m0 < F1_)              { width = F1_; cb = 0;          xin = x1; outp = out1; }
    else if (m0 < F1_ + F2_)   { width = F2_; cb = F1_;        xin = x2; outp = out2; }
    else                       { width = F3_; cb = F1_ + F2_;  xin = x3; outp = out3; }
    const int c0 = m0 + mo;                    // 8 contiguous channels
    const size_t rowoff = ((size_t)b * N_ + nc) * width + (c0 - cb);
    #pragma unroll
    for (int r = 0; r < 8; ++r) {
      const int c = c0 + r;
      const float scale = bn_g[c] * rsqrtf(bn_v[c] + 1e-5f);
      outp[rowoff + r] = (acc[m][r] - bn_m[c]) * scale + bn_b[c] + xin[rowoff + r];
    }
  }
}

// ---------------- launch ----------------
extern "C" void kernel_launch(void* const* d_in, const int* in_sizes, int n_in,
                              void* d_out, int out_size, void* d_ws, size_t ws_size,
                              hipStream_t stream) {
  const float* x1   = (const float*)d_in[0];
  const float* g1   = (const float*)d_in[1];
  const float* b1   = (const float*)d_in[2];
  const float* x2   = (const float*)d_in[3];
  const float* g2   = (const float*)d_in[4];
  const float* b2   = (const float*)d_in[5];
  const float* x3   = (const float*)d_in[6];
  const float* g3   = (const float*)d_in[7];
  const float* b3   = (const float*)d_in[8];
  const float* qkvw = (const float*)d_in[9];
  const float* dw3  = (const float*)d_in[10];
  const float* pw3  = (const float*)d_in[11];
  const float* dw5  = (const float*)d_in[12];
  const float* pw5  = (const float*)d_in[13];
  const float* pjw  = (const float*)d_in[14];
  const float* bng  = (const float*)d_in[15];
  const float* bnb  = (const float*)d_in[16];
  const float* bnm  = (const float*)d_in[17];
  const float* bnv  = (const float*)d_in[18];

  char* ws = (char*)d_ws;
  _Float16* xh   = (_Float16*)(ws + XH_OFF);
  _Float16* wq   = (_Float16*)(ws + WQ_OFF);
  _Float16* wp   = (_Float16*)(ws + WP_OFF);
  _Float16* qkvh = (_Float16*)(ws + QKV_OFF);
  _Float16* m3h  = (_Float16*)(ws + M3_OFF);
  _Float16* m5h  = (_Float16*)(ws + M5_OFF);
  float*    vk   = (float*)   (ws + VK_OFF);
  _Float16* atth = (_Float16*)(ws + ATT_OFF);

  float* out1 = (float*)d_out;
  float* out2 = out1 + (size_t)B_ * N_ * F1_;
  float* out3 = out2 + (size_t)B_ * N_ * F2_;

  cvt_weights<<<(C3_ * CIN_ + 255) / 256, 256, 0, stream>>>(qkvw, pjw, wq, wp);

  {
    int waves = B_ * N_ * 3;                  // one wave per (row, segment)
    int blocks = (waves + 7) / 8;
    ln_kernel<<<blocks, 256, 0, stream>>>(x1, g1, b1, x2, g2, b2, x3, g3, b3, xh);
  }

  qkv_gemm<<<B_ * 64, 128, 0, stream>>>(xh, wq, qkvh);      // 4 n-tiles per block

  msconv<<<B_ * NG_ * 16, 256, 0, stream>>>(qkvh, dw3, pw3, dw5, pw5, m3h, m5h);

  vk_reduce<<<B_ * NG_, 256, 0, stream>>>(qkvh, m3h, m5h, vk);

  att_apply<<<B_ * NG_ * 16, 256, 0, stream>>>(qkvh, m3h, m5h, vk, atth);

  proj_gemm<<<B_ * 64 * 10, 128, 0, stream>>>(atth, wp, bng, bnb, bnm, bnv,
                                              x1, x2, x3, out1, out2, out3);

  (void)in_sizes; (void)n_in; (void)out_size; (void)ws_size;
}
